// QANet_76622216561431
// MI455X (gfx1250) — matrix operand order repaired and checked by reading, not compiled, not run on hardware
//
#include <hip/hip_runtime.h>
#include <hip/hip_bf16.h>
#include <hip/hip_fp16.h>

// QANet context-query attention for MI455X (gfx1250, wave32, WMMA).
// B=16, D=128, LC=2048, LQ=256.

typedef __attribute__((ext_vector_type(16))) _Float16 v16h;
typedef __attribute__((ext_vector_type(8)))  float    v8f;
typedef __attribute__((ext_vector_type(4)))  int      v4i;

constexpr int kB  = 16;
constexpr int kD  = 128;
constexpr int kLC = 2048;
constexpr int kLQ = 256;
constexpr int kF  = 4 * kD;   // 512 features

// ---------------------------------------------------------------------------
// Async global->LDS staging (CDNA5 ASYNCcnt path) with safe fallback.
// ROCm clang-22 declares the builtin with generic `int4*` params:
//   (int4* gaddr, int4* ldsaddr, imm offset, imm cpol)
// ---------------------------------------------------------------------------
#if __has_builtin(__builtin_amdgcn_global_load_async_to_lds_b128)
#define HAVE_ASYNC_LDS 1
__device__ __forceinline__ void async_copy_b128(const void* g, void* l) {
  __builtin_amdgcn_global_load_async_to_lds_b128((v4i*)g, (v4i*)l, 0, 0);
}
#if __has_builtin(__builtin_amdgcn_s_wait_asynccnt)
#define WAIT_ASYNC(n) __builtin_amdgcn_s_wait_asynccnt(n)
#else
#define WAIT_ASYNC(n) asm volatile("s_wait_asynccnt " #n ::: "memory")
#endif
#else
#define HAVE_ASYNC_LDS 0
#define WAIT_ASYNC(n)
__device__ __forceinline__ void async_copy_b128(const void* g, void* l) {
  *(uint4*)l = *(const uint4*)g;
}
#endif

// ---------------------------------------------------------------------------
// WMMA fragment helpers (f16 16x16x32, f32 accumulate).
// A-matrix 16x32: lanes 0-15 rows M=0..15 hold K=0..15 (pairs per VGPR),
// lanes 16-31 same rows hold K=16..31.  K run is contiguous.
// ---------------------------------------------------------------------------
__device__ __forceinline__ v16h load_frag_A(const _Float16* src, int ld,
                                            int row0, int k0) {
  const int lane  = threadIdx.x & 31;
  const int m     = lane & 15;
  const int khalf = (lane >> 4) * 16;
  union { v16h h; unsigned int u[8]; } f;
  const _Float16* row = src + (size_t)(row0 + m) * ld + k0 + khalf;
#pragma unroll
  for (int v = 0; v < 8; ++v) {
    const int k = 2 * v;
    f.u[v] = *(const unsigned int*)(row + k);
  }
  return f.h;
}

// B-matrix 32x16 (KxN): lanes 0-15 hold column N=lane, K=0..15;
// lanes 16-31 hold same columns, K=16..31.  Source is laid out so each lane's
// K run is contiguous in memory (per-column row of length >= k0+32).
__device__ __forceinline__ v16h load_frag_B(const _Float16* src, int ld,
                                            int col0, int k0) {
  const int lane = threadIdx.x & 31;
  const int n    = lane & 15;
  const int kblk = (lane >> 4) * 16;
  union { v16h h; unsigned int u[8]; } f;
  const _Float16* row = src + (size_t)(col0 + n) * ld + k0 + kblk;
#pragma unroll
  for (int v = 0; v < 8; ++v) f.u[v] = *(const unsigned int*)(row + 2 * v);
  return f.h;
}

__device__ __forceinline__ v8f wmma_f16(v16h a, v16h b, v8f c) {
  return __builtin_amdgcn_wmma_f32_16x16x32_f16(
      /*neg_a=*/false, a, /*neg_b=*/false, b,
      /*c_mod=*/(short)0, c, /*reuse_a=*/false, /*reuse_b=*/false);
}

// ---------------------------------------------------------------------------
// 0a. Convert C side: Ct (LC,D) f16, Ct*w3 (LC,D) f16, C^T copy (D,LC) f16,
//     rC[i] = sum_d Ct[i,d]*w1[d].
// ---------------------------------------------------------------------------
__global__ void k_convC(const float* __restrict__ C, const float* __restrict__ w,
                        float* __restrict__ rC, _Float16* __restrict__ Ct,
                        _Float16* __restrict__ Ctw3, _Float16* __restrict__ CT) {
  const int idx = blockIdx.x * blockDim.x + threadIdx.x;   // b*LC + i
  if (idx >= kB * kLC) return;
  const int b = idx / kLC, i = idx % kLC;
  const float* Cb = C + (size_t)b * kD * kLC;
  float acc = 0.f;
#pragma unroll 4
  for (int d = 0; d < kD; ++d) {
    const float v = Cb[(size_t)d * kLC + i];
    acc += v * w[d];                                        // w1
    const _Float16 h = (_Float16)v;
    Ct  [((size_t)b * kLC + i) * kD + d] = h;
    Ctw3[((size_t)b * kLC + i) * kD + d] = (_Float16)(v * w[2 * kD + d]);
    CT  [((size_t)b * kD + d) * kLC + i] = h;
  }
  rC[idx] = acc;
}

// 0b. Convert Q side: Qt (LQ,D) f16, Q^T copy (D,LQ) f16, rQ = Qt*w2.
__global__ void k_convQ(const float* __restrict__ Q, const float* __restrict__ w,
                        float* __restrict__ rQ, _Float16* __restrict__ Qt,
                        _Float16* __restrict__ QT) {
  const int idx = blockIdx.x * blockDim.x + threadIdx.x;   // b*LQ + j
  if (idx >= kB * kLQ) return;
  const int b = idx / kLQ, j = idx % kLQ;
  const float* Qb = Q + (size_t)b * kD * kLQ;
  float acc = 0.f;
#pragma unroll 4
  for (int d = 0; d < kD; ++d) {
    const float v = Qb[(size_t)d * kLQ + j];
    acc += v * w[kD + d];                                   // w2
    const _Float16 h = (_Float16)v;
    Qt[((size_t)b * kLQ + j) * kD + d] = h;
    QT[((size_t)b * kD + d) * kLQ + j] = h;
  }
  rQ[idx] = acc;
}

// 0c. W_res (D,4D) f32 -> f16 (row-major, f contiguous per output channel d).
__global__ void k_convW(const float* __restrict__ W, _Float16* __restrict__ Wh) {
  const int i = blockIdx.x * blockDim.x + threadIdx.x;
  if (i < kD * kF) Wh[i] = (_Float16)W[i];
}

// ---------------------------------------------------------------------------
// 1. S[b,i,j] = sum_d (Ct*w3)[i,d] * Qt[j,d] + rC[i] + rQ[j].
//    One wave per 16x16 tile; K = D = 128 -> 4 WMMA steps.
// ---------------------------------------------------------------------------
__global__ void k_score(const _Float16* __restrict__ Ctw3,
                        const _Float16* __restrict__ Qt,
                        const float* __restrict__ rC, const float* __restrict__ rQ,
                        float* __restrict__ S) {
  const int wave = (blockIdx.x * blockDim.x + threadIdx.x) >> 5;
  constexpr int tilesPerB = (kLC / 16) * (kLQ / 16);       // 2048
  const int b  = wave / tilesPerB;
  const int t  = wave % tilesPerB;
  const int i0 = (t / (kLQ / 16)) * 16;
  const int j0 = (t % (kLQ / 16)) * 16;
  const _Float16* A  = Ctw3 + (size_t)b * kLC * kD;
  const _Float16* Bq = Qt   + (size_t)b * kLQ * kD;
  v8f acc = {};
#pragma unroll
  for (int k = 0; k < kD; k += 32)
    acc = wmma_f16(load_frag_A(A, kD, i0, k), load_frag_B(Bq, kD, j0, k), acc);

  const int lane = threadIdx.x & 31;
  const int n = lane & 15, mb = (lane >> 4) * 8;
  float* Sb = S + (size_t)b * kLC * kLQ;
  const float rq = rQ[b * kLQ + j0 + n];
#pragma unroll
  for (int v = 0; v < 8; ++v) {
    const int m = mb + v;
    Sb[(size_t)(i0 + m) * kLQ + (j0 + n)] = acc[v] + rC[b * kLC + i0 + m] + rq;
  }
}

// ---------------------------------------------------------------------------
// 2a. Row softmax (over j, qmask) -> S_row f16 (LC,LQ). One block per row.
// ---------------------------------------------------------------------------
__global__ void k_rowsoftmax(const float* __restrict__ S,
                             const int* __restrict__ qmask,
                             _Float16* __restrict__ Srow) {
  const int row = blockIdx.x;               // b*LC + i
  const int b = row / kLC;
  const int j = threadIdx.x;                // 0..255 == LQ
  __shared__ float red[kLQ];
  float v = S[(size_t)row * kLQ + j];
  if (qmask[b * kLQ + j] == 0) v = -1e30f;
  red[j] = v; __syncthreads();
  for (int s = kLQ / 2; s > 0; s >>= 1) {
    if (j < s) red[j] = fmaxf(red[j], red[j + s]);
    __syncthreads();
  }
  const float mx = red[0]; __syncthreads();
  const float e = __expf(v - mx);
  red[j] = e; __syncthreads();
  for (int s = kLQ / 2; s > 0; s >>= 1) {
    if (j < s) red[j] += red[j + s];
    __syncthreads();
  }
  Srow[(size_t)row * kLQ + j] = (_Float16)(e / red[0]);
}

// 2b. Column softmax (over i, cmask) -> S_col^T f16 stored (LQ,LC).
__global__ void k_colsoftmax(const float* __restrict__ S,
                             const int* __restrict__ cmask,
                             _Float16* __restrict__ ScolT) {
  const int col = blockIdx.x;               // b*LQ + j
  const int b = col / kLQ, j = col % kLQ;
  const int t = threadIdx.x;                // 256 threads x 8 elems
  __shared__ float red[256];
  const float* Sb = S + (size_t)b * kLC * kLQ + j;
  const int* cm = cmask + b * kLC;
  float vals[8];
  float mx = -3.4e38f;
#pragma unroll
  for (int e = 0; e < 8; ++e) {
    const int i = t + 256 * e;
    float v = Sb[(size_t)i * kLQ];
    if (cm[i] == 0) v = -1e30f;
    vals[e] = v; mx = fmaxf(mx, v);
  }
  red[t] = mx; __syncthreads();
  for (int s = 128; s > 0; s >>= 1) {
    if (t < s) red[t] = fmaxf(red[t], red[t + s]);
    __syncthreads();
  }
  mx = red[0]; __syncthreads();
  float sum = 0.f;
#pragma unroll
  for (int e = 0; e < 8; ++e) { vals[e] = __expf(vals[e] - mx); sum += vals[e]; }
  red[t] = sum; __syncthreads();
  for (int s = 128; s > 0; s >>= 1) {
    if (t < s) red[t] += red[t + s];
    __syncthreads();
  }
  sum = red[0];
  _Float16* out = ScolT + (size_t)col * kLC;
#pragma unroll
  for (int e = 0; e < 8; ++e) out[t + 256 * e] = (_Float16)(vals[e] / sum);
}

// ---------------------------------------------------------------------------
// 3. ScTC^T (D,LQ) = (S_col^T (LQ,LC) @ Ct (LC,D))^T.
//    Block-cooperative: one block = 32-row j-panel, 8 waves cover D=128.
//    S_col^T panel double-buffered in LDS via async global->LDS DMA
//    (ASYNCcnt), prefetch of next chunk overlaps WMMAs of current chunk.
//    Each wave: 2 accumulator tiles sharing one B fragment (2 WMMA / B fetch).
// ---------------------------------------------------------------------------
__global__ void __launch_bounds__(256)
k_sctc(const _Float16* __restrict__ ScolT, const _Float16* __restrict__ CT,
       _Float16* __restrict__ ScTCT) {
  constexpr int CH = 256;                   // K-chunk (halfs)
  constexpr int JR = 32;                    // j rows per block
  constexpr int NCH = kLC / CH;             // 8 chunks
  const int blk = blockIdx.x;               // b * (kLQ/JR) + j-panel
  const int b  = blk / (kLQ / JR);
  const int j0 = (blk % (kLQ / JR)) * JR;
  const int wave = threadIdx.x >> 5;
  const int lane = threadIdx.x & 31;
  const int d0 = wave * 16;
  const int tid = threadIdx.x;

  __shared__ _Float16 Ap[2][JR][CH];        // 2 x 16 KB double buffer

  const _Float16* A  = ScolT + (size_t)b * kLQ * kLC;
  const _Float16* Bc = CT    + (size_t)b * kD * kLC;

  auto stage = [&](int buf, int kc) {       // 4 x b128 per thread per chunk
#pragma unroll
    for (int s = 0; s < 4; ++s) {
      const int o = tid * 8 + s * 2048;     // half offset within 32x256 panel
      const int r = o >> 8, cl = o & (CH - 1);
      async_copy_b128(A + (size_t)(j0 + r) * kLC + kc + cl, &Ap[buf][r][cl]);
    }
  };

  v8f acc0 = {}, acc1 = {};
  stage(0, 0);
  for (int c = 0; c < NCH; ++c) {
    if (c + 1 < NCH) {
      stage((c + 1) & 1, (c + 1) * CH);
      __builtin_prefetch(Bc + (size_t)(d0 + (lane & 15)) * kLC + (c + 1) * CH,
                         0, 1);
      WAIT_ASYNC(4);                        // current chunk landed, next in flight
    } else {
      WAIT_ASYNC(0);
    }
    __syncthreads();
    const _Float16* Aps = &Ap[c & 1][0][0];
#pragma unroll
    for (int kk = 0; kk < CH; kk += 32) {
      const v16h bf = load_frag_B(Bc, kLC, d0, c * CH + kk);
      acc0 = wmma_f16(load_frag_A(Aps, CH, 0,  kk), bf, acc0);
      acc1 = wmma_f16(load_frag_A(Aps, CH, 16, kk), bf, acc1);
    }
    __syncthreads();
  }

  const int n = lane & 15, mb = (lane >> 4) * 8;
  _Float16* outp = ScTCT + (size_t)b * kD * kLQ + (size_t)(d0 + n) * kLQ;
#pragma unroll
  for (int v = 0; v < 8; ++v) {
    outp[j0 + mb + v]      = (_Float16)acc0[v];
    outp[j0 + 16 + mb + v] = (_Float16)acc1[v];
  }
}

// ---------------------------------------------------------------------------
// 4. Fused tail: per 16-row tile of i, 8 waves compute A = S_row@Qt and
//    Bm = S_row@ScTC (d-slice per wave), stage to LDS, build the 16x512
//    feature tile [Ct | A | Ct*A | Ct*Bm], then project with W_res f16
//    (16x16 output tile per wave, K=512), add bias, ReLU, store (B,D,LC).
// ---------------------------------------------------------------------------
__global__ void __launch_bounds__(256)
k_output(const _Float16* __restrict__ Srow, const _Float16* __restrict__ QT,
         const _Float16* __restrict__ ScTCT, const _Float16* __restrict__ Ct,
         const _Float16* __restrict__ Wres, const float* __restrict__ b_res,
         float* __restrict__ out) {
  const int blk = blockIdx.x;                 // b * (LC/16) + i-tile
  const int b  = blk / (kLC / 16);
  const int i0 = (blk % (kLC / 16)) * 16;
  const int wave = threadIdx.x >> 5;
  const int lane = threadIdx.x & 31;
  const int d0 = wave * 16;

  __shared__ _Float16 feats[16][kF];          // 16 KB
  __shared__ _Float16 Abuf[16][kD];           // 4 KB
  __shared__ _Float16 Bbuf[16][kD];           // 4 KB

  const _Float16* Arow = Srow  + (size_t)b * kLC * kLQ;   // (LC,LQ)
  const _Float16* Qb   = QT    + (size_t)b * kD * kLQ;    // (D,LQ)
  const _Float16* Sb   = ScTCT + (size_t)b * kD * kLQ;    // (D,LQ)

  v8f accA = {}, accB = {};
#pragma unroll
  for (int k = 0; k < kLQ; k += 32) {
    const v16h a = load_frag_A(Arow, kLQ, i0, k);
    accA = wmma_f16(a, load_frag_B(Qb, kLQ, d0, k), accA);
    accB = wmma_f16(a, load_frag_B(Sb, kLQ, d0, k), accB);
  }
  {
    const int n = lane & 15, mb = (lane >> 4) * 8;
#pragma unroll
    for (int v = 0; v < 8; ++v) {
      Abuf[mb + v][d0 + n] = (_Float16)accA[v];
      Bbuf[mb + v][d0 + n] = (_Float16)accB[v];
    }
  }
  __syncthreads();

  // Build the 16x512 feature tile.
  const _Float16* Ctb = Ct + ((size_t)b * kLC + i0) * kD;
  for (int e = threadIdx.x; e < 16 * kD; e += 256) {
    const int m = e >> 7, d = e & (kD - 1);
    const float ct = (float)Ctb[(size_t)m * kD + d];
    const float a  = (float)Abuf[m][d];
    const float bm = (float)Bbuf[m][d];
    feats[m][d]          = (_Float16)ct;
    feats[m][kD + d]     = (_Float16)a;
    feats[m][2 * kD + d] = (_Float16)(ct * a);
    feats[m][3 * kD + d] = (_Float16)(ct * bm);
  }
  __syncthreads();

  // Projection: out[b, d, i] = relu(sum_f feats[i,f] * W_res[d,f] + b_res[d]).
  v8f acc = {};
#pragma unroll
  for (int k = 0; k < kF; k += 32) {
    const v16h a  = load_frag_A(&feats[0][0], kF, 0, k);
    const v16h bf = load_frag_B(Wres, kF, d0, k);
    acc = wmma_f16(a, bf, acc);
  }
  const int n = lane & 15, mb = (lane >> 4) * 8;
  float* ob = out + (size_t)b * kD * kLC;
  const float bias = b_res[d0 + n];
#pragma unroll
  for (int v = 0; v < 8; ++v)
    ob[(size_t)(d0 + n) * kLC + (i0 + mb + v)] = fmaxf(acc[v] + bias, 0.f);
}

// ---------------------------------------------------------------------------
// Host launcher.
// ---------------------------------------------------------------------------
extern "C" void kernel_launch(void* const* d_in, const int* in_sizes, int n_in,
                              void* d_out, int out_size, void* d_ws, size_t ws_size,
                              hipStream_t stream) {
  const float* C     = (const float*)d_in[0];   // (B, D, LC)
  const float* Q     = (const float*)d_in[1];   // (B, D, LQ)
  const int*   cmask = (const int*)  d_in[2];   // (B, LC)
  const int*   qmask = (const int*)  d_in[3];   // (B, LQ)
  const float* w     = (const float*)d_in[4];   // (3D,)
  const float* W_res = (const float*)d_in[5];   // (D, 4D)
  const float* b_res = (const float*)d_in[6];   // (D,)
  float* out = (float*)d_out;                   // (B, D, LC)

  // Workspace layout (bytes).
  char* ws = (char*)d_ws;
  size_t off = 0;
  auto alloc = [&](size_t bytes) { char* p = ws + off; off += (bytes + 255) & ~(size_t)255; return p; };
  _Float16* Ctw3  = (_Float16*)alloc((size_t)kB * kLC * kD * 2);   // 8 MiB
  _Float16* Ct    = (_Float16*)alloc((size_t)kB * kLC * kD * 2);   // 8 MiB
  _Float16* CT    = (_Float16*)alloc((size_t)kB * kD * kLC * 2);   // 8 MiB
  _Float16* Qt    = (_Float16*)alloc((size_t)kB * kLQ * kD * 2);   // 1 MiB
  _Float16* QT    = (_Float16*)alloc((size_t)kB * kD * kLQ * 2);   // 1 MiB
  _Float16* ScTCT = (_Float16*)alloc((size_t)kB * kD * kLQ * 2);   // 1 MiB
  _Float16* Wh    = (_Float16*)alloc((size_t)kD * kF * 2);         // 128 KiB
  float*    rC    = (float*)   alloc((size_t)kB * kLC * 4);        // 128 KiB
  float*    rQ    = (float*)   alloc((size_t)kB * kLQ * 4);        // 16 KiB
  float*    S     = (float*)   alloc((size_t)kB * kLC * kLQ * 4);  // 32 MiB
  _Float16* Srow  = (_Float16*)alloc((size_t)kB * kLC * kLQ * 2);  // 16 MiB
  _Float16* ScolT = (_Float16*)alloc((size_t)kB * kLQ * kLC * 2);  // 16 MiB
  (void)ws_size; (void)n_in; (void)in_sizes; (void)out_size;

  // 0. Conversions / precompute.
  k_convC<<<(kB * kLC + 255) / 256, 256, 0, stream>>>(C, w, rC, Ct, Ctw3, CT);
  k_convQ<<<(kB * kLQ + 255) / 256, 256, 0, stream>>>(Q, w, rQ, Qt, QT);
  k_convW<<<(kD * kF + 255) / 256, 256, 0, stream>>>(W_res, Wh);

  // 1. Similarity matrix S (one wave per 16x16 tile).
  {
    const int waves = kB * (kLC / 16) * (kLQ / 16);                // 32768
    k_score<<<waves * 32 / 256, 256, 0, stream>>>(Ctw3, Qt, rC, rQ, S);
  }

  // 2. Softmaxes.
  k_rowsoftmax<<<kB * kLC, kLQ, 0, stream>>>(S, qmask, Srow);
  k_colsoftmax<<<kB * kLQ, 256, 0, stream>>>(S, cmask, ScolT);

  // 3. ScTC^T = (S_col^T @ Ct)^T  (async-LDS double-buffered GEMM).
  k_sctc<<<kB * (kLQ / 32), 256, 0, stream>>>(ScolT, CT, ScTCT);

  // 4. Fused A/Bm + features + projection + ReLU + transpose-out.
  k_output<<<kB * (kLC / 16), 256, 0, stream>>>(Srow, QT, ScTCT, Ct, Wh, b_res, out);
}
